// AE_14542759264437
// MI455X (gfx1250) — matrix operand order repaired and checked by reading, not compiled, not run on hardware
//
#include <hip/hip_runtime.h>
#include <hip/hip_bf16.h>
#include <math.h>

#define LVL   17
#define DD    256
#define MIXN  20
#define NINT  ((1 << LVL) - 1)   // 131071 internal nodes; leaves are >= NINT

typedef __attribute__((ext_vector_type(16))) __bf16 v16bf;
typedef __attribute__((ext_vector_type(8)))  float  v8f;

__device__ __forceinline__ v8f zero8() {
  v8f z;
#pragma unroll
  for (int i = 0; i < 8; ++i) z[i] = 0.f;
  return z;
}

__device__ __forceinline__ v8f wmma_bf16(v16bf a, v16bf b, v8f c) {
  // D = A(16x32 bf16) * B(32x16 bf16) + C(f32)
  return __builtin_amdgcn_wmma_f32_16x16x32_bf16(false, a, false, b, (short)0, c,
                                                 false, false);
}

// A-operand fragment from an f32 LDS tile (row stride `stride`), K-tile base kbase.
// A layout (matches packed B): lanes 0-15 hold M=lane, K in {0..15}; lanes 16-31
// hold M=lane-16, K in {16..31}; element e -> koff = half*16 + e.
__device__ __forceinline__ v16bf load_A(const float* lds, int stride, int kbase) {
  int lane = threadIdx.x & 31;
  int m = lane & 15, half = lane >> 4;
  v16bf a;
#pragma unroll
  for (int e = 0; e < 16; ++e) {
    int k = kbase + (half << 4) + e;
    a[e] = (__bf16)lds[m * stride + k];
  }
  return a;
}

// B-operand fragment straight from pre-packed weights (32B/lane -> 2x b128 loads)
__device__ __forceinline__ v16bf load_B(const __bf16* packed, int tile) {
  int lane = threadIdx.x & 31;
  return *(const v16bf*)(packed + ((size_t)tile * 32 + lane) * 16);
}

__device__ __forceinline__ float sigmf(float x) { return 1.f / (1.f + __expf(-x)); }

// ---------------------------------------------------------------------------
// Pack W (Nreal x Ksrc, row-major, used as x @ W^T) into bf16 WMMA-B tiles.
// B order used here (contiguous k, not AMD's table order): lane = N%16 (+16 for K-half 1), VGPR v holds K=2v,2v+1
// => element e -> K = kt*32 + (lane>>4)*16 + e.
// permute=1 remaps decoder-LSTM input order [p_f(8)|f_f(256)] -> [f_f|p_f|pad].
// ---------------------------------------------------------------------------
__global__ void pack_B(const float* __restrict__ src, __bf16* __restrict__ dst,
                       int Nreal, int Ksrc, int NT, int permute) {
  int tile = blockIdx.x;
  int kt = tile / NT, nt = tile - kt * NT;
  int lane = threadIdx.x;
  int n = nt * 16 + (lane & 15);
  int khalf = lane >> 4;
  v16bf out;
#pragma unroll
  for (int e = 0; e < 16; ++e) {
    int k = kt * 32 + khalf * 16 + e;
    int ks = k;
    if (permute) ks = (k < 256) ? (k + 8) : ((k < 264) ? (k - 256) : -1);
    float v = 0.f;
    if (n < Nreal && ks >= 0 && ks < Ksrc) v = src[(size_t)n * Ksrc + ks];
    out[e] = (__bf16)v;
  }
  *(v16bf*)(dst + ((size_t)tile * 32 + lane) * 16) = out;
}

// ---------------------------------------------------------------------------
// Encoder level: 16 LSTM cell rows per block (rows 2p/2p+1 = left/right child
// of parent p). g = x@Wih^T + h@Whh^T + b;  Feature_New[f] = [hl'+hr', cl'+cr'].
// ---------------------------------------------------------------------------
__global__ __launch_bounds__(256) void enc_level(
    const float* __restrict__ X, const float* __restrict__ Feature,
    const int* __restrict__ I_flat, const float* __restrict__ bih,
    const float* __restrict__ bhh, const __bf16* __restrict__ pWih,
    const __bf16* __restrict__ pWhh, float* __restrict__ FeatBuf,
    int Ioff, int cnt) {
  __shared__ float xs[16][32];
  __shared__ float hs[16][128];
  __shared__ float cs[16][128];
  __shared__ int sli[8], sri[8], sfi[8];

  int tid = threadIdx.x;
  if (tid < 8) {
    int p = blockIdx.x * 8 + tid;
    if (p >= cnt) p = cnt - 1;
    const int* row = I_flat + (size_t)(Ioff + p) * 3;
    sli[tid] = row[0]; sri[tid] = row[1]; sfi[tid] = row[2];
  }
  __syncthreads();

  for (int idx = tid; idx < 16 * 32; idx += 256) {
    int r = idx >> 5, c = idx & 31;
    int child = (r & 1) ? sri[r >> 1] : sli[r >> 1];
    xs[r][c] = (c < 8) ? X[(size_t)child * 8 + c] : 0.f;
  }
  for (int idx = tid; idx < 16 * 128; idx += 256) {
    int r = idx >> 7, c = idx & 127;
    int child = (r & 1) ? sri[r >> 1] : sli[r >> 1];
    const float* srcf = (child >= NINT) ? (Feature + (size_t)child * DD)
                                        : (FeatBuf + (size_t)child * DD);
    hs[r][c] = srcf[c];
    cs[r][c] = srcf[128 + c];
  }
  __syncthreads();

  int wave = tid >> 5, lane = tid & 31;
  int ln = lane & 15, halfm = lane >> 4;
  int col = wave * 16 + ln;  // column within each 128-wide gate

  v16bf ax = load_A(&xs[0][0], 32, 0);
  v16bf ah[4];
#pragma unroll
  for (int kt = 0; kt < 4; ++kt) ah[kt] = load_A(&hs[0][0], 128, kt * 32);

  v8f acc[4];
#pragma unroll
  for (int t = 0; t < 4; ++t) {           // t: gate part i/f/g/o
    int nt = wave + 8 * t;                // N-tile (0..31)
    v8f c = zero8();
    c = wmma_bf16(ax, load_B(pWih, nt), c);
#pragma unroll
    for (int kt = 0; kt < 4; ++kt)
      c = wmma_bf16(ah[kt], load_B(pWhh, kt * 32 + nt), c);
    acc[t] = c;
  }

  float b0 = bih[col] + bhh[col];
  float b1 = bih[128 + col] + bhh[128 + col];
  float b2 = bih[256 + col] + bhh[256 + col];
  float b3 = bih[384 + col] + bhh[384 + col];

  float hv[8], cv[8];
#pragma unroll
  for (int r = 0; r < 8; ++r) {
    int m = r + 8 * halfm;
    float ig = sigmf(acc[0][r] + b0);
    float fg = sigmf(acc[1][r] + b1);
    float gg = tanhf(acc[2][r] + b2);
    float og = sigmf(acc[3][r] + b3);
    float c2 = fg * cs[m][col] + ig * gg;
    cv[r] = c2;
    hv[r] = og * tanhf(c2);
  }
  // rows (2q,2q+1) of one parent land in accumulator rows r=2q,2q+1 of this lane
#pragma unroll
  for (int q = 0; q < 4; ++q) {
    int plocal = q + 4 * halfm;
    int p = blockIdx.x * 8 + plocal;
    if (p < cnt) {
      int f = sfi[plocal];
      FeatBuf[(size_t)f * DD + col]       = hv[2 * q] + hv[2 * q + 1];
      FeatBuf[(size_t)f * DD + 128 + col] = cv[2 * q] + cv[2 * q + 1];
    }
  }
}

// ---------------------------------------------------------------------------
// Mixture-density NLL for one point vs one 263-wide output row (in LDS).
// ---------------------------------------------------------------------------
__device__ float nll_eval(const float* pt, const float* y) {
  const float LOG2PI = 1.8378770664093453f;
  const float LOGSQRT2PI = 0.9189385332046727f;
  float mx = -1e30f;
  for (int k = 0; k < MIXN; ++k) mx = fmaxf(mx, y[k]);
  float se = 0.f;
  for (int k = 0; k < MIXN; ++k) se += __expf(y[k] - mx);
  float lsePi = mx + __logf(se);

  float dx = pt[0], dy = pt[1], da = pt[2], db = pt[3], dsv = pt[4];
  float m1 = -1e30f, s1 = 0.f, m2 = -1e30f, s2 = 0.f, m3 = -1e30f, s3 = 0.f;
  for (int k = 0; k < MIXN; ++k) {
    float lp = y[k] - lsePi;
    {  // (x,y) bivariate
      float lsx = y[60 + k], lsy = y[80 + k];
      float rho = tanhf(y[100 + k]);
      float zx = (dx - y[20 + k]) * __expf(-lsx);
      float zy = (dy - y[40 + k]) * __expf(-lsy);
      float omr = 1.f - rho * rho;
      float Z = zx * zx + zy * zy - 2.f * rho * zx * zy;
      float t = lp - Z / (2.f * omr) - LOG2PI - lsx - lsy - 0.5f * __logf(omr);
      if (t > m1) { s1 = s1 * __expf(m1 - t) + 1.f; m1 = t; } else s1 += __expf(t - m1);
    }
    {  // (a,b) bivariate
      float lsa = y[160 + k], lsb = y[180 + k];
      float rho = tanhf(y[200 + k]);
      float za = (da - y[120 + k]) * __expf(-lsa);
      float zb = (db - y[140 + k]) * __expf(-lsb);
      float omr = 1.f - rho * rho;
      float Z = za * za + zb * zb - 2.f * rho * za * zb;
      float t = lp - Z / (2.f * omr) - LOG2PI - lsa - lsb - 0.5f * __logf(omr);
      if (t > m2) { s2 = s2 * __expf(m2 - t) + 1.f; m2 = t; } else s2 += __expf(t - m2);
    }
    {  // scalar s
      float lss = y[240 + k];
      float zs = (dsv - y[220 + k]) * __expf(-lss);
      float t = lp - 0.5f * zs * zs - LOGSQRT2PI - lss;
      if (t > m3) { s3 = s3 * __expf(m3 - t) + 1.f; m3 = t; } else s3 += __expf(t - m3);
    }
  }
  float lxy = m1 + __logf(s1), lab = m2 + __logf(s2), lsl = m3 + __logf(s3);
  float q0 = y[260], q1 = y[261], q2 = y[262];
  float qm = fmaxf(q0, fmaxf(q1, q2));
  float lseq = qm + __logf(__expf(q0 - qm) + __expf(q1 - qm) + __expf(q2 - qm));
  float pen = -(pt[5] * (q0 - lseq) + pt[6] * (q1 - lseq) + pt[7] * (q2 - lseq));
  return -(lxy + lab + lsl) + pen;
}

// ---------------------------------------------------------------------------
// Decoder level: 16 nodes / block, 8 waves.
// LDS (f32, aliased): [ff 4096 | xp 512 | zc 4096] reused as yb(8704) after
// gates; zh 4096; ho 4096; co 4096; + 64 ints.  ~84 KB / workgroup.
// ---------------------------------------------------------------------------
__global__ __launch_bounds__(256) void dec_level(
    const float* __restrict__ X, const int* __restrict__ I_flat,
    const float* __restrict__ b_fch, const float* __restrict__ b_ihd,
    const float* __restrict__ b_hhd, const float* __restrict__ fc_b,
    const __bf16* __restrict__ pFch, const __bf16* __restrict__ pWihD,
    const __bf16* __restrict__ pWhhD, const __bf16* __restrict__ pFcW,
    float* __restrict__ FeatBuf, float* __restrict__ loss,
    int Ioff, int cnt, float scale) {
  extern __shared__ float smem[];
  float* ff = smem;              // 16x256  (aliased by yb later)
  float* xp = smem + 4096;       // 16x32
  float* zc = smem + 4608;       // 16x256
  float* yb = smem;              // 2x16x272 = 8704 (over ff|xp|zc)
  float* zh = smem + 8704;       // 16x256
  float* ho = smem + 12800;      // 16x256
  float* co = smem + 16896;      // 16x256
  int* sli = (int*)(smem + 20992);
  int* sri = sli + 16; int* sfi = sri + 16; int* ssw = sfi + 16;

  int tid = threadIdx.x;
  if (tid < 16) {
    int p = blockIdx.x * 16 + tid;
    if (p >= cnt) p = cnt - 1;
    const int* row = I_flat + (size_t)(Ioff + p) * 3;
    sli[tid] = row[0]; sri[tid] = row[1]; sfi[tid] = row[2];
  }
  __syncthreads();
  for (int idx = tid; idx < 16 * 256; idx += 256) {
    int r = idx >> 8, c = idx & 255;
    ff[idx] = FeatBuf[(size_t)sfi[r] * DD + c];
  }
  for (int idx = tid; idx < 16 * 32; idx += 256) {
    int r = idx >> 5, c = idx & 31;
    xp[idx] = (c < 8) ? X[(size_t)sfi[r] * 8 + c] : 0.f;
  }
  __syncthreads();

  int wave = tid >> 5, lane = tid & 31;
  int ln = lane & 15, halfm = lane >> 4;

  v16bf aff[8];
#pragma unroll
  for (int kt = 0; kt < 8; ++kt) aff[kt] = load_A(ff, 256, kt * 32);
  v16bf axp = load_A(xp, 32, 0);

  // ---- z = tanh(ff @ fc_h_W^T + b) ; h_f=z[:,:256] -> zh, c_f=z[:,256:] -> zc
#pragma unroll
  for (int t = 0; t < 4; ++t) {
    int nt = wave + 8 * t;                 // 0..31
    v8f c = zero8();
#pragma unroll
    for (int kt = 0; kt < 8; ++kt)
      c = wmma_bf16(aff[kt], load_B(pFch, kt * 32 + nt), c);
    int colg = nt * 16 + ln;
    float bb = b_fch[colg];
    float* dstz = (nt < 16) ? zh : zc;
    int colz = (nt < 16) ? colg : (colg - 256);
#pragma unroll
    for (int r = 0; r < 8; ++r) {
      int m = r + 8 * halfm;
      dstz[m * 256 + colz] = tanhf(c[r] + bb);
    }
  }
  __syncthreads();

  v16bf azh[8];
#pragma unroll
  for (int kt = 0; kt < 8; ++kt) azh[kt] = load_A(zh, 256, kt * 32);

  // ---- gates: g = [ff|x] @ Wihd'^T + h_f @ Whhd^T + b  (1024 wide) ----
#pragma unroll
  for (int gsel = 0; gsel < 2; ++gsel) {
    int ct = wave + 8 * gsel;              // within-gate tile 0..15
    int ccol = ct * 16 + ln;               // 0..255
    v8f a[4];
#pragma unroll
    for (int p = 0; p < 4; ++p) {          // i/f/g/o
      int nt = ct + 16 * p;                // 0..63
      v8f c = zero8();
#pragma unroll
      for (int kt = 0; kt < 8; ++kt)
        c = wmma_bf16(aff[kt], load_B(pWihD, kt * 64 + nt), c);
      c = wmma_bf16(axp, load_B(pWihD, 8 * 64 + nt), c);
#pragma unroll
      for (int kt = 0; kt < 8; ++kt)
        c = wmma_bf16(azh[kt], load_B(pWhhD, kt * 64 + nt), c);
      a[p] = c;
    }
    float b0 = b_ihd[ccol] + b_hhd[ccol];
    float b1 = b_ihd[256 + ccol] + b_hhd[256 + ccol];
    float b2 = b_ihd[512 + ccol] + b_hhd[512 + ccol];
    float b3 = b_ihd[768 + ccol] + b_hhd[768 + ccol];
#pragma unroll
    for (int r = 0; r < 8; ++r) {
      int m = r + 8 * halfm;
      float ig = sigmf(a[0][r] + b0);
      float fg = sigmf(a[1][r] + b1);
      float gg = tanhf(a[2][r] + b2);
      float og = sigmf(a[3][r] + b3);
      float c2 = fg * zc[m * 256 + ccol] + ig * gg;
      co[m * 256 + ccol] = c2;
      ho[m * 256 + ccol] = og * tanhf(c2);
    }
  }
  __syncthreads();   // ff/xp/zc now dead -> yb may overwrite

  // ---- y = h_{left,right} @ fc_W^T + fc_b  (34 tile-columns / 8 waves) ----
  for (int combo = wave; combo < 34; combo += 8) {   // wave-uniform trip count
    int side = combo & 1;                            // 0=left(h[0:128]) 1=right
    int nt = combo >> 1;                             // 0..16 (272 cols padded)
    v8f c = zero8();
#pragma unroll
    for (int kt = 0; kt < 4; ++kt) {
      v16bf af = load_A(ho, 256, side * 128 + kt * 32);
      c = wmma_bf16(af, load_B(pFcW, kt * 17 + nt), c);
    }
    int colg = nt * 16 + ln;
    float bb = (colg < 263) ? fc_b[colg] : 0.f;
#pragma unroll
    for (int r = 0; r < 8; ++r) {
      int m = r + 8 * halfm;
      yb[(side * 16 + m) * 272 + colg] = c[r] + bb;
    }
  }
  __syncthreads();

  // ---- per-node NLL + swap decision ----
  if (tid < 16) {
    int p = blockIdx.x * 16 + tid;
    int sw = 0;
    if (p < cnt) {
      const float* yl = yb + (size_t)tid * 272;
      const float* yr = yb + (size_t)(16 + tid) * 272;
      float ptl[8], ptr_[8];
      for (int j = 0; j < 8; ++j) {
        ptl[j]  = X[(size_t)sli[tid] * 8 + j];
        ptr_[j] = X[(size_t)sri[tid] * 8 + j];
      }
      float d = nll_eval(ptl, yl) + nll_eval(ptr_, yr);
      float s = nll_eval(ptl, yr) + nll_eval(ptr_, yl);
      sw = (s < d) ? 1 : 0;
      atomicAdd(loss, fminf(s, d) * scale);
    }
    ssw[tid] = sw;
  }
  __syncthreads();

  // ---- scatter child features (only internal children matter for Loss) ----
  for (int idx = tid; idx < 16 * 512; idx += 256) {
    int nd = idx >> 9;
    int p = blockIdx.x * 16 + nd;
    if (p >= cnt) continue;
    int cc = idx & 511;
    int isR = cc >> 8;
    int col = cc & 255;
    int child = isR ? sri[nd] : sli[nd];
    if (child >= NINT) continue;
    int side = ssw[nd] ^ isR;   // swap picks the other half
    float v = (col < 128) ? ho[nd * 256 + side * 128 + col]
                          : co[nd * 256 + side * 128 + (col - 128)];
    FeatBuf[(size_t)child * DD + col] = v;
  }
}

// ---------------------------------------------------------------------------
extern "C" void kernel_launch(void* const* d_in, const int* in_sizes, int n_in,
                              void* d_out, int out_size, void* d_ws, size_t ws_size,
                              hipStream_t stream) {
  const float* X       = (const float*)d_in[0];
  const float* Feature = (const float*)d_in[1];
  const int*   I_flat  = (const int*)d_in[2];
  const float* W_ih_e  = (const float*)d_in[3];
  const float* W_hh_e  = (const float*)d_in[4];
  const float* b_ih_e  = (const float*)d_in[5];
  const float* b_hh_e  = (const float*)d_in[6];
  const float* fc_h_W  = (const float*)d_in[7];
  const float* fc_h_b  = (const float*)d_in[8];
  const float* W_ih_d  = (const float*)d_in[9];
  const float* W_hh_d  = (const float*)d_in[10];
  const float* b_ih_d  = (const float*)d_in[11];
  const float* b_hh_d  = (const float*)d_in[12];
  const float* fc_W    = (const float*)d_in[13];
  const float* fc_b    = (const float*)d_in[14];
  (void)in_sizes; (void)n_in; (void)ws_size;

  char* ws = (char*)d_ws;
  size_t o = 0;
  auto take = [&](size_t bytes) {
    char* r = ws + o;
    o = (o + bytes + 255) & ~(size_t)255;
    return r;
  };
  __bf16* pWihE = (__bf16*)take((size_t)32  * 512 * 2);  // NT=32, KT=1
  __bf16* pWhhE = (__bf16*)take((size_t)128 * 512 * 2);  // NT=32, KT=4
  __bf16* pFch  = (__bf16*)take((size_t)256 * 512 * 2);  // NT=32, KT=8
  __bf16* pWihD = (__bf16*)take((size_t)576 * 512 * 2);  // NT=64, KT=9 (permuted)
  __bf16* pWhhD = (__bf16*)take((size_t)512 * 512 * 2);  // NT=64, KT=8
  __bf16* pFcW  = (__bf16*)take((size_t)68  * 512 * 2);  // NT=17, KT=4
  float* FeatBuf = (float*)take((size_t)NINT * DD * sizeof(float));

  pack_B<<<32,  32, 0, stream>>>(W_ih_e, pWihE, 512, 8,   32, 0);
  pack_B<<<128, 32, 0, stream>>>(W_hh_e, pWhhE, 512, 128, 32, 0);
  pack_B<<<256, 32, 0, stream>>>(fc_h_W, pFch,  512, 256, 32, 0);
  pack_B<<<576, 32, 0, stream>>>(W_ih_d, pWihD, 1024, 264, 64, 1);
  pack_B<<<512, 32, 0, stream>>>(W_hh_d, pWhhD, 1024, 256, 64, 0);
  pack_B<<<68,  32, 0, stream>>>(fc_W,   pFcW,  263, 128, 17, 0);

  int offs[LVL + 1]; offs[0] = 0;
  int cnts[LVL];
  for (int l = 0; l < LVL; ++l) {
    cnts[l] = 1 << (LVL - 1 - l);          // 65536 ... 1 (leaf-most first)
    offs[l + 1] = offs[l] + cnts[l];
  }

  // encoder: leaves -> root
  for (int l = 0; l < LVL; ++l) {
    int cnt = cnts[l];
    int blocks = (2 * cnt + 15) / 16;
    enc_level<<<blocks, 256, 0, stream>>>(X, Feature, I_flat, b_ih_e, b_hh_e,
                                          pWihE, pWhhE, FeatBuf, offs[l], cnt);
  }

  hipMemsetAsync(d_out, 0, (size_t)out_size * sizeof(float), stream);

  size_t smem = (size_t)(20992 + 64) * sizeof(float);  // ~84 KB
  // idempotent + deterministic: call every invocation (no static guards)
  hipFuncSetAttribute(reinterpret_cast<const void*>(dec_level),
                      hipFuncAttributeMaxDynamicSharedMemorySize, (int)smem);

  // decoder: root -> leaves
  for (int l = LVL - 1; l >= 0; --l) {
    int cnt = cnts[l];
    int blocks = (cnt + 15) / 16;
    float scale = 1.f / ((float)cnt * (float)LVL);
    dec_level<<<blocks, 256, smem, stream>>>(X, I_flat, fc_h_b, b_ih_d, b_hh_d,
                                             fc_b, pFch, pWihD, pWhhD, pFcW,
                                             FeatBuf, (float*)d_out, offs[l],
                                             cnt, scale);
  }
}
